// GedLayer_32504312496631
// MI455X (gfx1250) — matrix operand order repaired and checked by reading, not compiled, not run on hardware
//
#include <hip/hip_runtime.h>

typedef __attribute__((ext_vector_type(16))) _Float16 v16h;
typedef __attribute__((ext_vector_type(8)))  float    v8f;

namespace {
constexpr int NG   = 90;      // N_G == M_G
constexpr int NT   = 91;      // padded-by-one logical size
constexpr int NPAD = 96;      // 6 WMMA tiles of 16
constexpr int PITCH = 93;     // fp32 Sinkhorn pitch (odd -> conflict-free cols)
constexpr int NTHREADS = 512; // 16 wave32s
constexpr int NWAVES = NTHREADS / 32;
constexpr int NTILES = 36;    // 6x6 output tiles
constexpr int ELEMS  = NT * NT;   // 8281

// Shared-memory layout (bytes). Vs (fp32 Sinkhorn buffer) is dead after
// phase C, so Ps / A1p / A2p alias its space. Peak ~55.3 KB < 64 KB.
constexpr int VH_OFF  = 0;        // fp16 96x96 V            (18432 B)
constexpr int VS_OFF  = 18432;    // fp32 91x93 Sinkhorn     (33852 B)  [phases A-C]
constexpr int PS_OFF  = 18432;    // fp16 96x96 P            (18432 B)  [phase D, aliases Vs]
constexpr int A1_OFF  = 36864;    // u8 96x96 A1 (pad=255)   ( 9216 B)  [aliases Vs tail]
constexpr int A2_OFF  = 46080;    // u8 96x96 A2 (pad=0)     ( 9216 B)
constexpr int NC_OFF  = 55296;    // fp32 10x10 node costs   (  400 B)
constexpr int W_OFF   = 55696;    // fp32 3x3 W table        (   36 B)
constexpr int RS_OFF  = 55732;    // fp32 row scales         (  364 B)
constexpr int CS_OFF  = 56096;    // fp32 col scales         (  364 B)
constexpr int LIN_OFF = 56460;    // fp32 linear-term accum
constexpr int DQ_OFF  = 56464;    // fp32 diag-correction accum
constexpr int TS_OFF  = 56468;    // fp32 T[3][3] accums     (   36 B)
constexpr int NID_OFF = 56504;    // fp32 nodeInsDel
constexpr int SMEM_BYTES = 56512;
}

__global__ __launch_bounds__(NTHREADS)
void ged_kernel(const int* __restrict__ adj,     // (2, 90*90) int32
                const int* __restrict__ lab,     // (2, 90)    int32
                const float* __restrict__ nodew, // 46 fp32
                const float* __restrict__ edgew, // 2  fp32
                float* __restrict__ out)
{
  __shared__ __align__(16) unsigned char smem[SMEM_BYTES];
  _Float16* Vh = (_Float16*)(smem + VH_OFF);
  float*    Vs = (float*)(smem + VS_OFF);
  _Float16* Ps = (_Float16*)(smem + PS_OFF);
  unsigned char* A1p = smem + A1_OFF;
  unsigned char* A2p = smem + A2_OFF;
  float* ncost = (float*)(smem + NC_OFF);
  float* Wt    = (float*)(smem + W_OFF);
  float* rsc   = (float*)(smem + RS_OFF);
  float* csc   = (float*)(smem + CS_OFF);
  float* linS  = (float*)(smem + LIN_OFF);
  float* dqS   = (float*)(smem + DQ_OFF);
  float* Ts    = (float*)(smem + TS_OFF);
  float* nidS  = (float*)(smem + NID_OFF);

  const int tid  = threadIdx.x;
  const int lane = tid & 31;
  const int wv   = tid >> 5;
  const int lo   = lane & 15;
  const int hi   = lane >> 4;

  const int* adj2 = adj + NG * NG;
  const int* l1 = lab;
  const int* l2 = lab + NG;

  // ---- Phase A: tiny cost tables (serial on thread 0) ----
  if (tid == 0) {
    for (int k = 0; k < 100; ++k) ncost[k] = 0.f;
    int idx = 0;
    for (int i = 0; i < 10; ++i)
      for (int j = i + 1; j < 10; ++j) {
        float v = fmaxf(nodew[idx], 0.f);
        ncost[i * 10 + j] = v;
        ncost[j * 10 + i] = v;
        ++idx;
      }
    *nidS = fmaxf(nodew[45], 0.f);            // nodeInsDel = relu(cn[-1])
    float ce0 = fmaxf(edgew[0], 0.f);         // edge_costs off-diagonal
    float eid = fmaxf(edgew[1], 0.f);         // edgeInsDel
    // W[a1*3+a2] = edgeInsDel*xor(a1!=0,a2!=0) + edge_costs[a1-1][a2-1]
    Wt[0] = 0.f; Wt[1] = eid; Wt[2] = eid;
    Wt[3] = eid; Wt[4] = 0.f; Wt[5] = ce0;
    Wt[6] = eid; Wt[7] = ce0; Wt[8] = 0.f;
    *linS = 0.f; *dqS = 0.f;
    for (int k = 0; k < 9; ++k) Ts[k] = 0.f;
  }
  __syncthreads();
  const float nid = *nidS;

  // ---- Phase B: S0 = exp(-0.5 * Dg), then 10 Sinkhorn iterations ----
  for (int idx = tid; idx < ELEMS; idx += NTHREADS) {
    int i = idx / NT, p = idx % NT;
    float d;
    if (i < NG && p < NG)        d = ncost[l1[i] * 10 + l2[p]];
    else if (i == NG && p == NG) d = 0.f;
    else                         d = nid;
    Vs[i * PITCH + p] = expf(-0.5f * d);
  }
  __syncthreads();

  for (int it = 0; it < 10; ++it) {
    if (tid < NT) {
      float s = 0.f;
      for (int p = 0; p < NT; ++p) s += Vs[tid * PITCH + p];
      rsc[tid] = (tid == NG) ? 1.f : 1.f / s;
    }
    __syncthreads();
    for (int idx = tid; idx < ELEMS; idx += NTHREADS) {
      int i = idx / NT, p = idx % NT;
      Vs[i * PITCH + p] *= rsc[i];
    }
    __syncthreads();
    if (tid < NT) {
      float s = 0.f;
      for (int i = 0; i < NT; ++i) s += Vs[i * PITCH + tid];
      csc[tid] = (tid == NG) ? 1.f : 1.f / s;
    }
    __syncthreads();
    for (int idx = tid; idx < ELEMS; idx += NTHREADS) {
      int i = idx / NT, p = idx % NT;
      Vs[i * PITCH + p] *= csc[p];
    }
    __syncthreads();
  }

  // ---- Phase C: linear term, diagonal correction, fp16 padded copy ----
  float linPart = 0.f, dqPart = 0.f;
  for (int idx = tid; idx < ELEMS; idx += NTHREADS) {
    int i = idx / NT, p = idx % NT;
    float v = Vs[i * PITCH + p];
    float d;
    if (i < NG && p < NG)        d = ncost[l1[i] * 10 + l2[p]];
    else if (i == NG && p == NG) d = 0.f;
    else                         d = nid;
    linPart += d * v;                                  // c @ v
    int a1d = (i < NG) ? adj[i * NG + i]  : 0;         // F diagonal removal
    int a2d = (p < NG) ? adj2[p * NG + p] : 0;
    dqPart += v * v * Wt[a1d * 3 + a2d];
  }
  atomicAdd(linS, linPart);
  atomicAdd(dqS, dqPart);
  for (int idx = tid; idx < NPAD * NPAD; idx += NTHREADS) {
    int i = idx / NPAD, p = idx % NPAD;
    Vh[idx] = (i < NT && p < NT) ? (_Float16)Vs[i * PITCH + p] : (_Float16)0.f;
  }
  __syncthreads();   // all Vs reads done; its LDS space is recycled below
  for (int idx = tid; idx < NPAD * NPAD; idx += NTHREADS) {
    int i = idx / NPAD, j = idx % NPAD;
    // A1 pad (beyond 91x91) = 255: indicator never matches -> exact zeros.
    // Real border row/col 90 is value 0 (participates in the a1==0 mask).
    unsigned char v1, v2;
    if (i < NG && j < NG)      { v1 = (unsigned char)adj[i * NG + j];
                                 v2 = (unsigned char)adj2[i * NG + j]; }
    else if (i < NT && j < NT) { v1 = 0; v2 = 0; }
    else                       { v1 = 255; v2 = 0; }
    A1p[idx] = v1;
    A2p[idx] = v2;
  }
  __syncthreads();

  // ---- Phase D: per edge label a1: P = M1(a1) @ V, then fold
  //      G = P^T @ V against the A2 indicator mask. 16x16x32 f16 WMMA. ----
  for (int a1 = 0; a1 < 3; ++a1) {
    // P = M1 @ V  (96x96x96 padded; pad rows/cols come out exactly zero)
    for (int t = wv; t < NTILES; t += NWAVES) {
      int mt = t / 6, nt = t % 6;
      v8f acc = {};
      #pragma unroll
      for (int ks = 0; ks < 3; ++ks) {
        v16h a, b;
        #pragma unroll
        for (int e = 0; e < 16; ++e) {
          // A fragment: lane lo -> row M, K = e + 16*hi (matches B)
          int kk = ks * 32 + e + (hi << 4);
          int i = mt * 16 + lo;
          a[e] = ((int)A1p[i * NPAD + kk] == a1) ? (_Float16)1.f : (_Float16)0.f;
          // B fragment: lane lo -> col N, K = e + 16*hi
          int kb = ks * 32 + e + (hi << 4);
          b[e] = Vh[kb * NPAD + nt * 16 + lo];
        }
        acc = __builtin_amdgcn_wmma_f32_16x16x32_f16(false, a, false, b,
                                                     (short)0, acc, false, false);
      }
      #pragma unroll
      for (int r = 0; r < 8; ++r) {
        int i = mt * 16 + r + (hi << 3);
        int p = nt * 16 + lo;
        Ps[i * NPAD + p] = (_Float16)acc[r];
      }
    }
    __syncthreads();
    // G = P^T @ V; per-lane bucket accumulation, 3 LDS atomics per lane.
    float t0 = 0.f, t1 = 0.f, t2 = 0.f;
    for (int t = wv; t < NTILES; t += NWAVES) {
      int pt = t / 6, qt = t % 6;
      v8f acc = {};
      #pragma unroll
      for (int ks = 0; ks < 3; ++ks) {
        v16h a, b;
        #pragma unroll
        for (int e = 0; e < 16; ++e) {
          int kk = ks * 32 + e + (hi << 4);
          a[e] = Ps[kk * NPAD + pt * 16 + lo];       // (P^T)[p][i] = P[i][p]
          int kb = ks * 32 + e + (hi << 4);
          b[e] = Vh[kb * NPAD + qt * 16 + lo];
        }
        acc = __builtin_amdgcn_wmma_f32_16x16x32_f16(false, a, false, b,
                                                     (short)0, acc, false, false);
      }
      #pragma unroll
      for (int r = 0; r < 8; ++r) {
        // Pad entries of G are exactly 0.0f, so no bounds mask is needed.
        int p = pt * 16 + r + (hi << 3);
        int q = qt * 16 + lo;
        int a2 = (int)A2p[p * NPAD + q];
        float g = acc[r];
        t0 = (a2 == 0) ? t0 + g : t0;
        t1 = (a2 == 1) ? t1 + g : t1;
        t2 = (a2 == 2) ? t2 + g : t2;
      }
    }
    atomicAdd(&Ts[a1 * 3 + 0], t0);
    atomicAdd(&Ts[a1 * 3 + 1], t1);
    atomicAdd(&Ts[a1 * 3 + 2], t2);
    __syncthreads();
  }

  // ---- Phase E: ged = 0.5*(Q - diag) + lin ----
  if (tid == 0) {
    float Q = 0.f;
    for (int k = 0; k < 9; ++k) Q += Wt[k] * Ts[k];
    out[0] = 0.5f * (Q - *dqS) + *linS;
  }
}

extern "C" void kernel_launch(void* const* d_in, const int* in_sizes, int n_in,
                              void* d_out, int out_size, void* d_ws, size_t ws_size,
                              hipStream_t stream) {
  (void)in_sizes; (void)n_in; (void)out_size; (void)d_ws; (void)ws_size;
  // inputs: 0=graph (unused), 1=adjacenceMatrix, 2=graphCard (unused),
  //         3=labels, 4=node_weighs, 5=edge_weighs
  const int*   adj = (const int*)d_in[1];
  const int*   lab = (const int*)d_in[3];
  const float* nw  = (const float*)d_in[4];
  const float* ew  = (const float*)d_in[5];
  ged_kernel<<<1, NTHREADS, 0, stream>>>(adj, lab, nw, ew, (float*)d_out);
}
